// LSTMDecoder_63960652972565
// MI455X (gfx1250) — compile-verified, matrix operand order checked by reading, not run on hardware
//
#include <hip/hip_runtime.h>

#define B_SZ   64
#define S_SZ   512
#define VOCAB  32000
#define EMBED  512
#define HIDDEN 1024
#define ENC    1024
#define ATTN   256

typedef __attribute__((ext_vector_type(16))) __bf16   v16bf;
typedef __attribute__((ext_vector_type(8)))  float    v8f;
typedef __attribute__((ext_vector_type(8)))  unsigned v8u;

// Pack two f32 into two RNE-rounded bf16 in one dword:
// bias-add for round-to-nearest-even, then v_perm_b32 grabs both high halves.
__device__ __forceinline__ unsigned pack2_bf16(float f0, float f1) {
  unsigned u0 = __builtin_bit_cast(unsigned, f0);
  unsigned u1 = __builtin_bit_cast(unsigned, f1);
  u0 += 0x7FFFu + ((u0 >> 16) & 1u);
  u1 += 0x7FFFu + ((u1 >> 16) & 1u);
  // D[15:0] = u0[31:16] (sel bytes 2,3 of S1), D[31:16] = u1[31:16] (bytes 6,7 of S0)
  return __builtin_amdgcn_perm(u1, u0, 0x07060302u);
}

// K offset inside a 16x32 bf16 fragment for element pair v (ISA 7.12.2):
// VGPR 0-3 hold K {0..7 | 8..15} (halves), VGPR 4-7 hold K {16..23 | 24..31}
__device__ __forceinline__ int frag_koff(int v, int half) {
  return ((v >> 2) << 4) + (half << 3) + ((v & 3) << 1);
}

// Build a bf16 WMMA fragment from a row-major f32 matrix (ld = leading dim).
// For A: row0 = tile M origin (lane -> M). For B taken from a weight stored
// [N,K] row-major, pass row0 = tile N origin (lane -> N): B(k,n) = W[n*ld+k].
// All (row*ld + k0 + koff) offsets are even -> float2 loads are 8B aligned.
__device__ __forceinline__ v16bf load_frag_rowK(const float* __restrict__ src,
                                                int ld, int row0, int k0, int lane) {
  const int lo = lane & 15, half = lane >> 4;
  const float* rp = src + (size_t)(row0 + lo) * ld + k0;
  v8u pk;
#pragma unroll
  for (int v = 0; v < 8; ++v) {
    const float2 p = *(const float2*)(rp + frag_koff(v, half));
    pk[v] = pack2_bf16(p.x, p.y);
  }
  return __builtin_bit_cast(v16bf, pk);
}

__device__ __forceinline__ v8f wmma_bf16(v16bf a, v16bf b, v8f c) {
  return __builtin_amdgcn_wmma_f32_16x16x32_bf16(
      /*neg_a=*/false, a, /*neg_b=*/false, b,
      /*c_mod=*/(short)0, c, /*reuse_a=*/false, /*reuse_b=*/false);
}

// ---------------------------------------------------------------- embedding
__global__ void k_embed(const int* __restrict__ tok, const float* __restrict__ table,
                        float* __restrict__ lstm_in, float* __restrict__ cat2) {
  int idx = blockIdx.x * blockDim.x + threadIdx.x;
  if (idx >= B_SZ * EMBED) return;
  int m = idx / EMBED, k = idx - m * EMBED;
  float e = table[(size_t)tok[m] * EMBED + k];
  lstm_in[m * (EMBED + ENC) + k] = e;                      // lstm_in[:, :E]
  cat2[m * (HIDDEN + ENC + EMBED) + HIDDEN + ENC + k] = e; // readout cat[:, H+ENC:]
}

// ------------------------------------------------- dec_proj = h @ W_dec^T
__global__ void __launch_bounds__(128)
k_decproj(const float* __restrict__ h_top, const float* __restrict__ W_dec,
          float* __restrict__ dec_proj) {
  int wave = threadIdx.x >> 5, lane = threadIdx.x & 31;
  int gid = blockIdx.x * 4 + wave;     // 64 tiles = 4 (M) x 16 (N)
  int m0 = (gid & 3) * 16, n0 = (gid >> 2) * 16;
  v8f acc = {};
  for (int k0 = 0; k0 < HIDDEN; k0 += 32) {
    v16bf a = load_frag_rowK(h_top, HIDDEN, m0, k0, lane);
    v16bf b = load_frag_rowK(W_dec, HIDDEN, n0, k0, lane);
    acc = wmma_bf16(a, b, acc);
  }
  int lo = lane & 15, half = lane >> 4, n = n0 + lo;
#pragma unroll
  for (int j = 0; j < 8; ++j)
    dec_proj[(m0 + j + 8 * half) * ATTN + n] = acc[j];
}

// --- scores[b,s] = v . tanh(dec_proj[b] + enc_outputs[b,s] @ W_enc^T) + mask
// One block per 16 flattened (b,s) rows; 16 waves each own a 16-wide N slice.
__global__ void __launch_bounds__(512)
k_scores(const float* __restrict__ enc, const float* __restrict__ W_enc,
         const float* __restrict__ dec_proj, const float* __restrict__ v_attn,
         const unsigned char* __restrict__ src_mask, float* __restrict__ scores) {
  __shared__ float Atile[16 * 32];
  __shared__ float partials[16][16];
  int tid = threadIdx.x;
  int wave = tid >> 5, lane = tid & 31;
  int lo = lane & 15, half = lane >> 4;
  int r0 = blockIdx.x * 16;        // flattened row tile (one b: 512 % 16 == 0)
  int bb = r0 / S_SZ;
  int n0 = wave * 16, n = n0 + lo;
  float dp = dec_proj[bb * ATTN + n];
  float vn = v_attn[n];
  int arow = tid >> 5, acol = tid & 31;          // 16x32 cooperative stage
  const float* Aglob = enc + (size_t)(r0 + arow) * ENC;
  v8f acc = {};
  for (int k0 = 0; k0 < ENC; k0 += 32) {
    __syncthreads();
    Atile[arow * 32 + acol] = Aglob[k0 + acol];
    if (k0 + 32 < ENC) __builtin_prefetch(&Aglob[k0 + 32 + acol], 0, 1);
    __syncthreads();
    v16bf a = load_frag_rowK(Atile, 32, 0, 0, lane);
    v16bf b = load_frag_rowK(W_enc, ENC, n0, k0, lane);
    acc = wmma_bf16(a, b, acc);
  }
  // reduce tanh(dp + ep)*v over this wave's 16 columns, per M row
#pragma unroll
  for (int j = 0; j < 8; ++j) {
    float p = tanhf(dp + acc[j]) * vn;
    p += __shfl_xor(p, 1, 32);
    p += __shfl_xor(p, 2, 32);
    p += __shfl_xor(p, 4, 32);
    p += __shfl_xor(p, 8, 32);
    if (lo == 0) partials[wave][j + 8 * half] = p;
  }
  __syncthreads();
  if (tid < 16) {                   // sum the 16 wave slices -> full 256-dot
    float t = 0.f;
#pragma unroll
    for (int w = 0; w < 16; ++w) t += partials[w][tid];
    int r = r0 + tid;
    int s = r - bb * S_SZ;
    scores[r] = (src_mask[bb * S_SZ + s] != 0) ? t : -1e9f;
  }
}

// --------------------------- softmax over S + context = weights @ enc[b]
__global__ void __launch_bounds__(256)
k_softmax_ctx(const float* __restrict__ scores, const float* __restrict__ enc,
              float* __restrict__ weights_out, float* __restrict__ lstm_in,
              float* __restrict__ cat2) {
  __shared__ float w[S_SZ];
  __shared__ float red[256];
  int bb = blockIdx.x, t = threadIdx.x;
  float s0 = scores[bb * S_SZ + t];
  float s1 = scores[bb * S_SZ + t + 256];
  red[t] = fmaxf(s0, s1);
  __syncthreads();
  for (int st = 128; st > 0; st >>= 1) {
    if (t < st) red[t] = fmaxf(red[t], red[t + st]);
    __syncthreads();
  }
  float mx = red[0];
  __syncthreads();
  float e0 = __expf(s0 - mx), e1 = __expf(s1 - mx);
  red[t] = e0 + e1;
  __syncthreads();
  for (int st = 128; st > 0; st >>= 1) {
    if (t < st) red[t] += red[t + st];
    __syncthreads();
  }
  float inv = 1.f / red[0];
  w[t] = e0 * inv;
  w[t + 256] = e1 * inv;
  weights_out[bb * S_SZ + t]       = w[t];
  weights_out[bb * S_SZ + t + 256] = w[t + 256];
  __syncthreads();
  float a0 = 0.f, a1 = 0.f, a2 = 0.f, a3 = 0.f;
  const float* eb = enc + (size_t)bb * S_SZ * ENC;
  for (int s = 0; s < S_SZ; ++s) {
    float wv = w[s];
    const float* row = eb + (size_t)s * ENC;
    a0 += wv * row[t];
    a1 += wv * row[t + 256];
    a2 += wv * row[t + 512];
    a3 += wv * row[t + 768];
  }
  float* li = lstm_in + bb * (EMBED + ENC) + EMBED;             // lstm_in[:, E:]
  float* c2 = cat2 + bb * (HIDDEN + ENC + EMBED) + HIDDEN;      // cat[:, H:H+ENC]
  li[t] = a0; li[t + 256] = a1; li[t + 512] = a2; li[t + 768] = a3;
  c2[t] = a0; c2[t + 256] = a1; c2[t + 512] = a2; c2[t + 768] = a3;
}

// ------- gates = lstm_in @ W_ih^T + h @ W_hh^T + b_ih + b_hh  (M=64,N=4096)
__global__ void __launch_bounds__(128)
k_gates(const float* __restrict__ lstm_in, const float* __restrict__ W_ih,
        const float* __restrict__ h_top, const float* __restrict__ W_hh,
        const float* __restrict__ b_ih, const float* __restrict__ b_hh,
        float* __restrict__ gates) {
  int wave = threadIdx.x >> 5, lane = threadIdx.x & 31;
  int gid = blockIdx.x * 4 + wave;          // 1024 tiles = 4 x 256
  int m0 = (gid & 3) * 16, n0 = (gid >> 2) * 16;
  v8f acc = {};
  for (int k0 = 0; k0 < EMBED + ENC; k0 += 32) {
    v16bf a = load_frag_rowK(lstm_in, EMBED + ENC, m0, k0, lane);
    v16bf b = load_frag_rowK(W_ih, EMBED + ENC, n0, k0, lane);
    acc = wmma_bf16(a, b, acc);
  }
  for (int k0 = 0; k0 < HIDDEN; k0 += 32) {
    v16bf a = load_frag_rowK(h_top, HIDDEN, m0, k0, lane);
    v16bf b = load_frag_rowK(W_hh, HIDDEN, n0, k0, lane);
    acc = wmma_bf16(a, b, acc);
  }
  int lo = lane & 15, half = lane >> 4, n = n0 + lo;
  float bias = b_ih[n] + b_hh[n];
#pragma unroll
  for (int j = 0; j < 8; ++j)
    gates[(m0 + j + 8 * half) * (4 * HIDDEN) + n] = acc[j] + bias;
}

// -------------------------------------------------- LSTM cell elementwise
__global__ void k_lstmcell(const float* __restrict__ gates, const float* __restrict__ c0,
                           float* __restrict__ newh_out, float* __restrict__ newc_out,
                           float* __restrict__ cat2) {
  int idx = blockIdx.x * blockDim.x + threadIdx.x;
  if (idx >= B_SZ * HIDDEN) return;
  int m = idx / HIDDEN, n = idx - m * HIDDEN;
  const float* g = gates + (size_t)m * 4 * HIDDEN;
  float ig = g[n], fg = g[HIDDEN + n], gg = g[2 * HIDDEN + n], og = g[3 * HIDDEN + n];
  float si = 1.f / (1.f + __expf(-ig));
  float sf = 1.f / (1.f + __expf(-fg));
  float so = 1.f / (1.f + __expf(-og));
  float nc = sf * c0[idx] + si * tanhf(gg);
  float nh = so * tanhf(nc);
  newc_out[idx] = nc;
  newh_out[idx] = nh;
  cat2[m * (HIDDEN + ENC + EMBED) + n] = nh;   // cat[:, :H]
}

// ------------------- t = tanh(cat @ W_out^T + b_out)   (M=64,N=512,K=2560)
__global__ void __launch_bounds__(128)
k_readout(const float* __restrict__ cat2, const float* __restrict__ W_out,
          const float* __restrict__ b_out, float* __restrict__ tout) {
  int wave = threadIdx.x >> 5, lane = threadIdx.x & 31;
  int gid = blockIdx.x * 4 + wave;          // 128 tiles = 4 x 32
  int m0 = (gid & 3) * 16, n0 = (gid >> 2) * 16;
  const int K = HIDDEN + ENC + EMBED;
  v8f acc = {};
  for (int k0 = 0; k0 < K; k0 += 32) {
    v16bf a = load_frag_rowK(cat2, K, m0, k0, lane);
    v16bf b = load_frag_rowK(W_out, K, n0, k0, lane);
    acc = wmma_bf16(a, b, acc);
  }
  int lo = lane & 15, half = lane >> 4, n = n0 + lo;
  float bias = b_out[n];
#pragma unroll
  for (int j = 0; j < 8; ++j)
    tout[(m0 + j + 8 * half) * EMBED + n] = tanhf(acc[j] + bias);
}

// ------------------------- logits = t @ emb_table^T  (M=64,N=32000,K=512)
__global__ void __launch_bounds__(128)
k_logits(const float* __restrict__ t, const float* __restrict__ emb_table,
         float* __restrict__ logits) {
  int wave = threadIdx.x >> 5, lane = threadIdx.x & 31;
  int gid = blockIdx.x * 4 + wave;          // 8000 tiles = 4 x 2000
  int m0 = (gid & 3) * 16, n0 = (gid >> 2) * 16;
  v8f acc = {};
  for (int k0 = 0; k0 < EMBED; k0 += 32) {
    v16bf a = load_frag_rowK(t, EMBED, m0, k0, lane);
    v16bf b = load_frag_rowK(emb_table, EMBED, n0, k0, lane);
    acc = wmma_bf16(a, b, acc);
  }
  int lo = lane & 15, half = lane >> 4, n = n0 + lo;
#pragma unroll
  for (int j = 0; j < 8; ++j)
    logits[(size_t)(m0 + j + 8 * half) * VOCAB + n] = acc[j];
}

extern "C" void kernel_launch(void* const* d_in, const int* in_sizes, int n_in,
                              void* d_out, int out_size, void* d_ws, size_t ws_size,
                              hipStream_t stream) {
  const int*   prev_token = (const int*)  d_in[0];
  const float* h0         = (const float*)d_in[1];   // [1,64,1024] -> top layer
  const float* c0         = (const float*)d_in[2];
  const float* enc        = (const float*)d_in[3];
  const unsigned char* mask = (const unsigned char*)d_in[4];
  const float* emb_table  = (const float*)d_in[5];
  const float* W_enc      = (const float*)d_in[6];
  const float* W_dec      = (const float*)d_in[7];
  const float* v_attn     = (const float*)d_in[8];
  const float* W_ih       = (const float*)d_in[9];
  const float* W_hh       = (const float*)d_in[10];
  const float* b_ih       = (const float*)d_in[11];
  const float* b_hh       = (const float*)d_in[12];
  const float* W_out      = (const float*)d_in[13];
  const float* b_out      = (const float*)d_in[14];

  float* out     = (float*)d_out;                 // logits, new_h, new_c, weights
  float* logits  = out;
  float* newh    = out + (size_t)B_SZ * VOCAB;
  float* newc    = newh + B_SZ * HIDDEN;
  float* weights = newc + B_SZ * HIDDEN;

  float* ws       = (float*)d_ws;                 // ~2.4 MB total
  float* dec_proj = ws;  ws += B_SZ * ATTN;
  float* scores   = ws;  ws += B_SZ * S_SZ;
  float* lstm_in  = ws;  ws += B_SZ * (EMBED + ENC);
  float* cat2     = ws;  ws += B_SZ * (HIDDEN + ENC + EMBED);
  float* gates    = ws;  ws += B_SZ * 4 * HIDDEN;
  float* tbuf     = ws;  ws += B_SZ * EMBED;

  k_embed<<<(B_SZ * EMBED + 255) / 256, 256, 0, stream>>>(prev_token, emb_table,
                                                          lstm_in, cat2);
  k_decproj<<<16, 128, 0, stream>>>(h0, W_dec, dec_proj);
  k_scores<<<(B_SZ * S_SZ) / 16, 512, 0, stream>>>(enc, W_enc, dec_proj, v_attn,
                                                   mask, scores);
  k_softmax_ctx<<<B_SZ, 256, 0, stream>>>(scores, enc, weights, lstm_in, cat2);
  k_gates<<<256, 128, 0, stream>>>(lstm_in, W_ih, h0, W_hh, b_ih, b_hh, gates);
  k_lstmcell<<<(B_SZ * HIDDEN + 255) / 256, 256, 0, stream>>>(gates, c0, newh,
                                                              newc, cat2);
  k_readout<<<32, 128, 0, stream>>>(cat2, W_out, b_out, tbuf);
  k_logits<<<2000, 128, 0, stream>>>(tbuf, emb_table, logits);
}
